// WanSelfAttention_15848429322608
// MI455X (gfx1250) — matrix operand order repaired and checked by reading, not compiled, not run on hardware
//
#include <hip/hip_runtime.h>
#include <cstdint>
#include <cstddef>

#define L_SEQ       4096
#define C_DIM       1536
#define D_HEAD      128
#define N_HEAD      12
#define ROPE_HALF   64
#define NT_PER_WAVE 12      // 96 column tiles / 8 waves
#define WAVES_PER_BLOCK 8

typedef _Float16 f16;
typedef __attribute__((ext_vector_type(16))) _Float16 v16h;
typedef __attribute__((ext_vector_type(8)))  float    v8f;
typedef __attribute__((ext_vector_type(8)))  uint32_t v8u;
typedef __attribute__((ext_vector_type(4)))  uint32_t v4u;
typedef __attribute__((ext_vector_type(4)))  int      v4i;

#if __has_builtin(__builtin_amdgcn_global_load_async_to_lds_b128)
#define HAS_ASYNC_COPY 1
#else
#define HAS_ASYNC_COPY 0
#endif

union FragU { v8u u; v16h h; };

static __device__ __forceinline__ v8f vzero8() {
  v8f z = {0.f,0.f,0.f,0.f,0.f,0.f,0.f,0.f};
  return z;
}

// 16-byte global -> LDS copy: async DMA (ASYNCcnt) when available, else staged.
static __device__ __forceinline__ void cp16(const f16* __restrict__ g,
                                            f16* __restrict__ l) {
#if HAS_ASYNC_COPY
  __builtin_amdgcn_global_load_async_to_lds_b128((v4i*)g, (v4i*)l, 0, 0);
#else
  v4u t = *(const v4u*)g;
  *(v4u*)l = t;
#endif
}

static __device__ __forceinline__ void async_wait_all() {
#if HAS_ASYNC_COPY
#if __has_builtin(__builtin_amdgcn_s_wait_asynccnt)
  __builtin_amdgcn_s_wait_asynccnt(0);
#else
  asm volatile("s_wait_asynccnt 0x0" ::: "memory");
#endif
#endif
}

// A-fragment (16x32 f16, M rows x K): lane m = lane&15, half = lane>>4.
// VGPR v<4 : K = 8*half + 2v ; v>=4 : K = 16 + 8*half + 2(v-4).  (ISA 7.12.2)
static __device__ __forceinline__ v16h load_a_frag(const f16* __restrict__ A, int lda) {
  const int lane = threadIdx.x & 31;
  const int hh = lane >> 4;
  const int m  = lane & 15;
  FragU f;
#pragma unroll
  for (int v = 0; v < 8; ++v) {
    const int k = (v < 4) ? (8 * hh + 2 * v) : (16 + 8 * hh + 2 * (v - 4));
    f.u[v] = *(const uint32_t*)(A + (size_t)m * lda + k);
  }
  return f.h;
}

// B-fragment (32x16 f16, K x N).  B[k][n] = Bmat[n*ldb + k].
// lane n = lane&15, half = lane>>4 ; VGPR v<4 : K = 8*half + 2v ; v>=4 : K = 16 + 8*half + 2(v-4).
static __device__ __forceinline__ v16h load_b_frag(const f16* __restrict__ Bmat, int ldb) {
  const int lane = threadIdx.x & 31;
  const int hh = lane >> 4;
  const int n  = lane & 15;
  FragU f;
#pragma unroll
  for (int v = 0; v < 8; ++v) {
    const int k = (v < 4) ? (8 * hh + 2 * v) : (16 + 8 * hh + 2 * (v - 4));
    f.u[v] = *(const uint32_t*)(Bmat + (size_t)n * ldb + k);
  }
  return f.h;
}

static __device__ __forceinline__ v8f wmma_f16(v16h a, v16h b, v8f c) {
  return __builtin_amdgcn_wmma_f32_16x16x32_f16(false, a, false, b, (short)0, c,
                                                false, false);
}

__global__ void cvt_f32_f16_kernel(const float* __restrict__ in,
                                   f16* __restrict__ out, int n) {
  int i = blockIdx.x * blockDim.x + threadIdx.x;
  if (i < n) out[i] = (f16)in[i];
}

// ---------------------------------------------------------------------------
// QKV projection: grid (L/16, 3). Block = 256 threads = 8 waves.
// Wave w owns 12 N-tiles -> full 1536 columns of one 16-token row block.
// Fused: bias, RMSNorm over C, gain, RoPE; writes f16 Q/K [h][t][d], V^T [h][d][t].
// ---------------------------------------------------------------------------
__global__ __launch_bounds__(256) void proj_qkv_kernel(
    const f16* __restrict__ xh,
    const f16* __restrict__ wq, const f16* __restrict__ wk, const f16* __restrict__ wv,
    const float* __restrict__ bq, const float* __restrict__ bk, const float* __restrict__ bv,
    const float* __restrict__ gq, const float* __restrict__ gk,
    const float* __restrict__ fc, const float* __restrict__ fs,
    f16* __restrict__ Qh, f16* __restrict__ Kh, f16* __restrict__ Vt) {
  const int proj = blockIdx.y;                       // 0=q 1=k 2=v (block-uniform)
  const f16*   W    = (proj == 0) ? wq : (proj == 1) ? wk : wv;
  const float* bias = (proj == 0) ? bq : (proj == 1) ? bk : bv;
  const float* g    = (proj == 0) ? gq : gk;

  const int qb   = blockIdx.x * 16;
  const int w    = threadIdx.x >> 5;
  const int lane = threadIdx.x & 31;
  const int hh   = lane >> 4;
  const int nl   = lane & 15;

  __shared__ float rowsq[16];

  v8f acc[NT_PER_WAVE];
#pragma unroll
  for (int t = 0; t < NT_PER_WAVE; ++t) acc[t] = vzero8();

  const f16* Abase = xh + (size_t)qb * C_DIM;
  for (int kc = 0; kc < C_DIM; kc += 32) {
    __builtin_prefetch((const void*)(Abase + kc + 32), 0, 1);
    v16h a = load_a_frag(Abase + kc, C_DIM);
#pragma unroll
    for (int t = 0; t < NT_PER_WAVE; ++t) {
      const int nt = (w * NT_PER_WAVE + t) * 16;
      v16h b = load_b_frag(W + (size_t)nt * C_DIM + kc, C_DIM);
      acc[t] = wmma_f16(a, b, acc[t]);
    }
  }

  // bias (per output column)
#pragma unroll
  for (int t = 0; t < NT_PER_WAVE; ++t) {
    const int c = (w * NT_PER_WAVE + t) * 16 + nl;
    const float bb = bias[c];
#pragma unroll
    for (int r = 0; r < 8; ++r) acc[t][r] += bb;
  }

  if (proj < 2) {
    // ---- RMSNorm over the full C row -------------------------------------
    if (threadIdx.x < 16) rowsq[threadIdx.x] = 0.f;
    __syncthreads();
    float ss[8];
#pragma unroll
    for (int r = 0; r < 8; ++r) ss[r] = 0.f;
#pragma unroll
    for (int t = 0; t < NT_PER_WAVE; ++t)
#pragma unroll
      for (int r = 0; r < 8; ++r) ss[r] += acc[t][r] * acc[t][r];
#pragma unroll
    for (int r = 0; r < 8; ++r) {
#pragma unroll
      for (int mk = 1; mk < 16; mk <<= 1) ss[r] += __shfl_xor(ss[r], mk, 32);
    }
    if (nl == 0) {
#pragma unroll
      for (int r = 0; r < 8; ++r) atomicAdd(&rowsq[r + 8 * hh], ss[r]);
    }
    __syncthreads();
    float rms[8];
#pragma unroll
    for (int r = 0; r < 8; ++r)
      rms[r] = rsqrtf(rowsq[r + 8 * hh] * (1.0f / C_DIM) + 1e-6f);

    // ---- gain + RoPE + store f16 [h][t][d] -------------------------------
    f16* dst = (proj == 0) ? Qh : Kh;
#pragma unroll
    for (int t = 0; t < NT_PER_WAVE; ++t) {
      const int c   = (w * NT_PER_WAVE + t) * 16 + nl;
      const float gv = g[c];
      const int p   = (c >> 1) & (ROPE_HALF - 1);
      const int odd = c & 1;
      const int hd  = c >> 7;            // head
      const int d   = c & (D_HEAD - 1);
#pragma unroll
      for (int r = 0; r < 8; ++r) {
        const int l = qb + r + 8 * hh;
        float val  = acc[t][r] * rms[r] * gv;
        float part = __shfl_xor(val, 1, 32);     // adjacent column = lane^1
        const float cv = fc[l * ROPE_HALF + p];
        const float sv = fs[l * ROPE_HALF + p];
        const float o = odd ? (part * sv + val * cv)   // im' = re*s + im*c
                            : (val * cv - part * sv);  // re' = re*c - im*s
        dst[((size_t)hd * L_SEQ + l) * D_HEAD + d] = (f16)o;
      }
    }
  } else {
    // ---- V: store transposed f16 [h][d][t] -------------------------------
#pragma unroll
    for (int t = 0; t < NT_PER_WAVE; ++t) {
      const int c  = (w * NT_PER_WAVE + t) * 16 + nl;
      const int hd = c >> 7;
      const int d  = c & (D_HEAD - 1);
#pragma unroll
      for (int r = 0; r < 8; ++r) {
        const int l = qb + r + 8 * hh;
        Vt[((size_t)hd * D_HEAD + d) * L_SEQ + l] = (f16)acc[t][r];
      }
    }
  }
}

// ---------------------------------------------------------------------------
// Flash attention: grid (L/(16*8), heads). Block = 8 waves, one 16-query tile
// each, all on the SAME head so the 32-key K/V tiles are shared through LDS
// (8x reduction in L2 traffic). Tiles are double-buffered and filled with
// async global->LDS DMA (ASYNCcnt), overlapped with WMMA compute.
// ---------------------------------------------------------------------------
__global__ __launch_bounds__(256) void attn_kernel(
    const f16* __restrict__ Qh, const f16* __restrict__ Kh,
    const f16* __restrict__ Vt, const int* __restrict__ seq_lens,
    f16* __restrict__ Oh) {
  const int h    = blockIdx.y;
  const int w    = threadIdx.x >> 5;
  const int lane = threadIdx.x & 31;
  const int hh   = lane >> 4;
  const int nl   = lane & 15;
  const int tid  = threadIdx.x;
  const int qb   = (blockIdx.x * WAVES_PER_BLOCK + w) * 16;
  const int slen = seq_lens[0];
  const float scale = 0.088388347648318447f;   // 1/sqrt(128)

  __shared__ f16 sK[2][32 * D_HEAD];             // 2 x 8KB : 32 keys x 128 d
  __shared__ f16 sV[2][D_HEAD * 32];             // 2 x 8KB : 128 d x 32 keys
  __shared__ f16 shp[WAVES_PER_BLOCK][16 * 32];  // per-wave P staging, 1 KB each
  f16* myp = shp[w];

  const f16* Qbase = Qh + ((size_t)h * L_SEQ + qb) * D_HEAD;
  const f16* Kbase = Kh + (size_t)h * L_SEQ * D_HEAD;
  const f16* Vbase = Vt + (size_t)h * D_HEAD * L_SEQ;

  // Cooperative stage of one 32-key K/V tile into LDS buffer `buf`.
  auto stage_kv = [&](int kb, int buf) {
    const f16* kg = Kbase + (size_t)kb * D_HEAD;     // 8KB contiguous
#pragma unroll
    for (int c = 0; c < 2; ++c) {
      const int chunk = tid + c * 256;               // 512 x 16B
      cp16(kg + chunk * 8, &sK[buf][chunk * 8]);
    }
#pragma unroll
    for (int c = 0; c < 2; ++c) {
      const int chunk = tid + c * 256;               // 512 x 16B
      const int row = chunk >> 2;                    // d   in [0,128)
      const int col = (chunk & 3) * 8;               // key in {0,8,16,24}
      cp16(Vbase + (size_t)row * L_SEQ + kb + col, &sV[buf][row * 32 + col]);
    }
  };

  v16h qf[4];                       // Q 16x128 resident as 4 A-fragments
#pragma unroll
  for (int i = 0; i < 4; ++i) qf[i] = load_a_frag(Qbase + 32 * i, D_HEAD);

  v8f oacc[8];
#pragma unroll
  for (int t = 0; t < 8; ++t) oacc[t] = vzero8();

  float mrow[8], lrow[8];
#pragma unroll
  for (int r = 0; r < 8; ++r) { mrow[r] = -1e30f; lrow[r] = 0.f; }

  stage_kv(0, 0);
  async_wait_all();
  __syncthreads();

  int buf = 0;
  for (int kb = 0; kb < L_SEQ; kb += 32) {
    if (kb + 32 < L_SEQ) stage_kv(kb + 32, buf ^ 1);   // overlap next tile DMA

    const f16* kt    = sK[buf];
    const f16* vtile = sV[buf];

    // ---- S = Q K^T for 32 keys (two 16x16 tiles), K from LDS -------------
    v8f s0 = vzero8(), s1 = vzero8();
#pragma unroll
    for (int i = 0; i < 4; ++i) {
      v16h b0 = load_b_frag(kt + 32 * i, D_HEAD);
      s0 = wmma_f16(qf[i], b0, s0);
      v16h b1 = load_b_frag(kt + 16 * D_HEAD + 32 * i, D_HEAD);
      s1 = wmma_f16(qf[i], b1, s1);
    }
    const int k0 = kb + nl, k1 = kb + 16 + nl;
#pragma unroll
    for (int r = 0; r < 8; ++r) {
      s0[r] = (k0 < slen) ? s0[r] * scale : -1e30f;
      s1[r] = (k1 < slen) ? s1[r] * scale : -1e30f;
    }
    // ---- online softmax --------------------------------------------------
    float tmax[8];
#pragma unroll
    for (int r = 0; r < 8; ++r) tmax[r] = fmaxf(s0[r], s1[r]);
#pragma unroll
    for (int r = 0; r < 8; ++r) {
#pragma unroll
      for (int mk = 1; mk < 16; mk <<= 1)
        tmax[r] = fmaxf(tmax[r], __shfl_xor(tmax[r], mk, 32));
    }
    float al[8];
#pragma unroll
    for (int r = 0; r < 8; ++r) {
      const float mn = fmaxf(mrow[r], tmax[r]);
      al[r] = __expf(mrow[r] - mn);
      mrow[r] = mn;
    }
    float psum[8];
#pragma unroll
    for (int r = 0; r < 8; ++r) {
      const float p0 = __expf(s0[r] - mrow[r]);
      const float p1 = __expf(s1[r] - mrow[r]);
      s0[r] = p0; s1[r] = p1;
      psum[r] = p0 + p1;
    }
#pragma unroll
    for (int r = 0; r < 8; ++r) {
#pragma unroll
      for (int mk = 1; mk < 16; mk <<= 1) psum[r] += __shfl_xor(psum[r], mk, 32);
      lrow[r] = lrow[r] * al[r] + psum[r];
    }
#pragma unroll
    for (int t = 0; t < 8; ++t)
#pragma unroll
      for (int r = 0; r < 8; ++r) oacc[t][r] *= al[r];

    // ---- C-layout P -> A-fragment via per-wave LDS (intra-wave in-order) -
#pragma unroll
    for (int r = 0; r < 8; ++r) {
      const int m = r + 8 * hh;
      myp[m * 32 + nl]      = (f16)s0[r];
      myp[m * 32 + 16 + nl] = (f16)s1[r];
    }
    v16h pa;
    {
      FragU f;
#pragma unroll
      for (int v = 0; v < 8; ++v) {
        const int k = (v < 4) ? (8 * hh + 2 * v) : (16 + 8 * hh + 2 * (v - 4));
        f.u[v] = *(const uint32_t*)(myp + nl * 32 + k);
      }
      pa = f.h;
    }
    // ---- O += P V, V from LDS (ldb = 32) ---------------------------------
#pragma unroll
    for (int t = 0; t < 8; ++t) {
      v16h vb = load_b_frag(vtile + t * 16 * 32, 32);
      oacc[t] = wmma_f16(pa, vb, oacc[t]);
    }

    async_wait_all();     // my async DMAs for the next buffer are done
    __syncthreads();      // everyone's are done; buf^1 now valid, buf reusable
    buf ^= 1;
  }

  float invl[8];
#pragma unroll
  for (int r = 0; r < 8; ++r) invl[r] = (lrow[r] > 0.f) ? 1.f / lrow[r] : 0.f;
#pragma unroll
  for (int t = 0; t < 8; ++t) {
    const int c = h * D_HEAD + t * 16 + nl;
#pragma unroll
    for (int r = 0; r < 8; ++r) {
      const int row = qb + r + 8 * hh;
      Oh[(size_t)row * C_DIM + c] = (f16)(oacc[t][r] * invl[r]);
    }
  }
}

// ---------------------------------------------------------------------------
// Output projection: out = attn @ Wo^T + bo, f32 result.
// ---------------------------------------------------------------------------
__global__ __launch_bounds__(256) void proj_out_kernel(
    const f16* __restrict__ Oh, const f16* __restrict__ Wo,
    const float* __restrict__ bo, float* __restrict__ out) {
  const int qb   = blockIdx.x * 16;
  const int w    = threadIdx.x >> 5;
  const int lane = threadIdx.x & 31;
  const int hh   = lane >> 4;
  const int nl   = lane & 15;

  v8f acc[NT_PER_WAVE];
#pragma unroll
  for (int t = 0; t < NT_PER_WAVE; ++t) acc[t] = vzero8();

  const f16* Abase = Oh + (size_t)qb * C_DIM;
  for (int kc = 0; kc < C_DIM; kc += 32) {
    __builtin_prefetch((const void*)(Abase + kc + 32), 0, 1);
    v16h a = load_a_frag(Abase + kc, C_DIM);
#pragma unroll
    for (int t = 0; t < NT_PER_WAVE; ++t) {
      const int nt = (w * NT_PER_WAVE + t) * 16;
      v16h b = load_b_frag(Wo + (size_t)nt * C_DIM + kc, C_DIM);
      acc[t] = wmma_f16(a, b, acc[t]);
    }
  }
#pragma unroll
  for (int t = 0; t < NT_PER_WAVE; ++t) {
    const int c = (w * NT_PER_WAVE + t) * 16 + nl;
    const float bb = bo[c];
#pragma unroll
    for (int r = 0; r < 8; ++r) {
      const int l = qb + r + 8 * hh;
      out[(size_t)l * C_DIM + c] = acc[t][r] + bb;
    }
  }
}

// ---------------------------------------------------------------------------
extern "C" void kernel_launch(void* const* d_in, const int* in_sizes, int n_in,
                              void* d_out, int out_size, void* d_ws, size_t ws_size,
                              hipStream_t stream) {
  const float* x        = (const float*)d_in[0];
  const int*   seq_lens = (const int*)d_in[1];
  // d_in[2] = grid_sizes (unused by reference math)
  const float* fc = (const float*)d_in[3];
  const float* fs = (const float*)d_in[4];
  const float* Wq = (const float*)d_in[5];
  const float* bq = (const float*)d_in[6];
  const float* Wk = (const float*)d_in[7];
  const float* bk = (const float*)d_in[8];
  const float* Wv = (const float*)d_in[9];
  const float* bv = (const float*)d_in[10];
  const float* Wo = (const float*)d_in[11];
  const float* bo = (const float*)d_in[12];
  const float* gq = (const float*)d_in[13];
  const float* gk = (const float*)d_in[14];
  float* out = (float*)d_out;

  char* ws = (char*)d_ws;
  size_t off = 0;
  auto carve = [&](size_t bytes) -> void* {
    void* p = (void*)(ws + off);
    off += (bytes + 255) & ~(size_t)255;
    return p;
  };
  const size_t xh_bytes  = (size_t)L_SEQ * C_DIM * sizeof(f16);
  const size_t w_bytes   = (size_t)C_DIM * C_DIM * sizeof(f16);
  const size_t qkv_bytes = (size_t)N_HEAD * L_SEQ * D_HEAD * sizeof(f16);
  f16* xh  = (f16*)carve(xh_bytes);   // x in f16; later reused as attn output
  f16* wqh = (f16*)carve(w_bytes);
  f16* wkh = (f16*)carve(w_bytes);
  f16* wvh = (f16*)carve(w_bytes);
  f16* woh = (f16*)carve(w_bytes);
  f16* qh  = (f16*)carve(qkv_bytes);  // [h][t][d]
  f16* kh  = (f16*)carve(qkv_bytes);  // [h][t][d]
  f16* vt  = (f16*)carve(qkv_bytes);  // [h][d][t]

  const int nx = L_SEQ * C_DIM;
  const int nw = C_DIM * C_DIM;
  cvt_f32_f16_kernel<<<(nx + 255) / 256, 256, 0, stream>>>(x,  xh,  nx);
  cvt_f32_f16_kernel<<<(nw + 255) / 256, 256, 0, stream>>>(Wq, wqh, nw);
  cvt_f32_f16_kernel<<<(nw + 255) / 256, 256, 0, stream>>>(Wk, wkh, nw);
  cvt_f32_f16_kernel<<<(nw + 255) / 256, 256, 0, stream>>>(Wv, wvh, nw);
  cvt_f32_f16_kernel<<<(nw + 255) / 256, 256, 0, stream>>>(Wo, woh, nw);

  dim3 gproj(L_SEQ / 16, 3);
  proj_qkv_kernel<<<gproj, 256, 0, stream>>>(xh, wqh, wkh, wvh, bq, bk, bv,
                                             gq, gk, fc, fs, qh, kh, vt);

  dim3 gattn(L_SEQ / (16 * WAVES_PER_BLOCK), N_HEAD);
  attn_kernel<<<gattn, 256, 0, stream>>>(qh, kh, vt, seq_lens, xh /* attn out */);

  proj_out_kernel<<<L_SEQ / 16, 256, 0, stream>>>(xh, woh, bo, out);

  (void)in_sizes; (void)n_in; (void)out_size; (void)ws_size;
}
